// HieraFormer_83751862272310
// MI455X (gfx1250) — matrix operand order repaired and checked by reading, not compiled, not run on hardware
//
#include <hip/hip_runtime.h>
#include <hip/hip_bf16.h>

// HieraFormer block for MI455X (gfx1250, wave32, WMMA).
// B=2, C=256, N=2048, H=4, DK=64.

#define BB 2
#define CC 256
#define NN 2048
#define HH 4
#define DKC 64
#define ESTRIDE 2056   // 2048 + 8 floats pad; 2056*4 bytes is 16B-aligned

typedef __attribute__((ext_vector_type(16))) _Float16 v16h;
typedef __attribute__((ext_vector_type(8)))  _Float16 v8h;
typedef __attribute__((ext_vector_type(8)))  float    v8f;

// A-operand (16x32 f16): lanes0-15 row M=lane hold K=0..15;
// lanes16-31 row M=lane-16 hold K=16..31 (matches load_b16).
static __device__ inline v16h load_a16(const _Float16* row, int k0, int laneHi) {
  const int base = k0 + (laneHi ? 16 : 0);
  v8h a0 = *(const v8h*)(row + base);
  v8h a1 = *(const v8h*)(row + base + 8);
  v16h r;
#pragma unroll
  for (int i = 0; i < 8; ++i) { r[i] = a0[i]; r[8 + i] = a1[i]; }
  return r;
}

// B-operand (32x16 f16): lane holds one column (16 contiguous K values);
// lanes0-15 K=k0..k0+15, lanes16-31 K=k0+16..k0+31.
static __device__ inline v16h load_b16(const _Float16* colrow, int k0, int laneHi) {
  return *(const v16h*)(colrow + k0 + (laneHi ? 16 : 0));
}

static __device__ inline float wave_sum(float v) {
#pragma unroll
  for (int m = 16; m; m >>= 1) v += __shfl_xor(v, m, 32);
  return v;
}
static __device__ inline float wave_max(float v) {
#pragma unroll
  for (int m = 16; m; m >>= 1) v = fmaxf(v, __shfl_xor(v, m, 32));
  return v;
}

// ---------------------------------------------------------------------------
// Kernel 0: x [B,C,N] -> xt f32 [B,N,C] (residual) + xh f16 [B,N,C] (GEMM A)
// ---------------------------------------------------------------------------
__global__ void hf_transpose(const float* __restrict__ x,
                             float* __restrict__ xt,
                             _Float16* __restrict__ xh) {
  __shared__ float tile[32][33];
  const int n0 = blockIdx.x * 32, c0 = blockIdx.y * 32, b = blockIdx.z;
  for (int i = threadIdx.y; i < 32; i += 8)
    tile[i][threadIdx.x] = x[((size_t)b * CC + c0 + i) * NN + n0 + threadIdx.x];
  __syncthreads();
  for (int i = threadIdx.y; i < 32; i += 8) {
    float v = tile[threadIdx.x][i];          // x[c0+tx][n0+i]
    size_t idx = ((size_t)b * NN + n0 + i) * CC + c0 + threadIdx.x;
    xt[idx] = v;
    xh[idx] = (_Float16)v;
  }
}

__global__ void hf_cvt_f16(const float* __restrict__ s, _Float16* __restrict__ d, int n) {
  int i = blockIdx.x * blockDim.x + threadIdx.x;
  if (i < n) d[i] = (_Float16)s[i];
}

// ---------------------------------------------------------------------------
// Kernel 1: QKV projections. Grid (N/16, B), 256 threads (8 waves).
// 48 output tiles (3 matrices x 16 col-tiles) -> 6 tiles/wave, K-loop 256/32.
// Q gets the 1/16 scale (1/sqrt(DK) * 1/2 for entmax z). V stored transposed.
// ---------------------------------------------------------------------------
__global__ void __launch_bounds__(256) hf_qkv(
    const _Float16* __restrict__ xh,
    const _Float16* __restrict__ wqh, const _Float16* __restrict__ wkh,
    const _Float16* __restrict__ wvh,
    const float* __restrict__ bq, const float* __restrict__ bk,
    const float* __restrict__ bv,
    _Float16* __restrict__ qbuf, _Float16* __restrict__ kbuf,
    _Float16* __restrict__ vtbuf) {
  const int lane = threadIdx.x & 31, wid = threadIdx.x >> 5;
  const int hiL = lane >> 4, ln = lane & 15;
  const int rowbase = blockIdx.x * 16, b = blockIdx.y;

  const _Float16* arow = xh + ((size_t)b * NN + rowbase + ln) * CC;
  v16h A[8];
#pragma unroll
  for (int c = 0; c < 8; ++c) A[c] = load_a16(arow, c * 32, hiL);

  for (int t = 0; t < 6; ++t) {
    const int tt = wid * 6 + t;
    const int m = tt >> 4, ot = tt & 15;
    const _Float16* w = (m == 0) ? wqh : (m == 1) ? wkh : wvh;
    const float* bias = (m == 0) ? bq : (m == 1) ? bk : bv;
    const int col = ot * 16 + ln;                // output channel for this lane
    const _Float16* wrow = w + (size_t)col * CC; // y = x @ W^T: B col = W row
    v8f acc = {};
#pragma unroll
    for (int c = 0; c < 8; ++c) {
      v16h Bm = load_b16(wrow, c * 32, hiL);
      acc = __builtin_amdgcn_wmma_f32_16x16x32_f16(false, A[c], false, Bm,
                                                   (short)0, acc, false, false);
    }
    const float bia = bias[col];
    const int h = col >> 6, dk = col & 63;
    if (m == 2) {
      // V -> Vt[B,H,DK,N]; per C/D layout lane covers 8 consecutive M (=n)
      v8h pk;
#pragma unroll
      for (int i = 0; i < 8; ++i) pk[i] = (_Float16)(acc[i] + bia);
      const int nb = rowbase + 8 * hiL;
      *(v8h*)(vtbuf + (((size_t)b * HH + h) * DKC + dk) * NN + nb) = pk;
    } else {
      _Float16* dst = (m == 0) ? qbuf : kbuf;
      const float sc = (m == 0) ? (1.0f / 16.0f) : 1.0f;
#pragma unroll
      for (int i = 0; i < 8; ++i) {
        const int n = rowbase + i + 8 * hiL;
        dst[(((size_t)b * HH + h) * NN + n) * DKC + dk] =
            (_Float16)((acc[i] + bia) * sc);
      }
    }
  }
}

// ---------------------------------------------------------------------------
// Kernel 2: fused attention. Grid (N/16, H, B), 256 threads, ~136.5 KB LDS.
// Phase 1: E[16][2048] = z = (QK^T)/16 via WMMA, fp32 in LDS.
// Phase 2: entmax15 per row. Each wave owns 2 rows; the row lives in 64 VGPRs
//          per lane. tau* is the unique root of sum(max(z-tau,0)^2)=1 in
//          [zmax-1, zmax] (monotone) -> 28 bisection steps == sort-based tau.
//          Result p is written back IN PLACE as packed f16 (rows are
//          wave-private; barrier before phase 3).
// Phase 3: O = attn * V via WMMA; A fragments come straight from packed-f16
//          LDS rows with two ds_load_b128 per lane. Cross-wave LDS reduce.
// ---------------------------------------------------------------------------
__global__ void __launch_bounds__(256) hf_attn(
    const _Float16* __restrict__ qbuf, const _Float16* __restrict__ kbuf,
    const _Float16* __restrict__ vtbuf, float* __restrict__ obuf) {
  extern __shared__ float smem[];
  float* E = smem;                   // 16 * ESTRIDE floats
  float* scr = smem + 16 * ESTRIDE;  // 8 * 256 floats (partial O tiles)
  const int lane = threadIdx.x & 31, wid = threadIdx.x >> 5;
  const int hiL = lane >> 4, ln = lane & 15;
  const int ntile = blockIdx.x, h = blockIdx.y, b = blockIdx.z;
  const size_t bh = (size_t)b * HH + h;

  // ---- Phase 1: energy strip -------------------------------------------
  const _Float16* qrow = qbuf + (bh * NN + ntile * 16 + ln) * DKC;
  const v16h aq0 = load_a16(qrow, 0, hiL);
  const v16h aq1 = load_a16(qrow, 32, hiL);
  const _Float16* kb = kbuf + bh * NN * DKC;
  for (int t = wid; t < NN / 16; t += 8) {
    const _Float16* krow = kb + (size_t)(t * 16 + ln) * DKC;  // B col = K row
    v16h b0 = load_b16(krow, 0, hiL);
    v16h b1 = load_b16(krow, 32, hiL);
    v8f acc = {};
    acc = __builtin_amdgcn_wmma_f32_16x16x32_f16(false, aq0, false, b0,
                                                 (short)0, acc, false, false);
    acc = __builtin_amdgcn_wmma_f32_16x16x32_f16(false, aq1, false, b1,
                                                 (short)0, acc, false, false);
#pragma unroll
    for (int i = 0; i < 8; ++i)
      E[(i + 8 * hiL) * ESTRIDE + t * 16 + ln] = acc[i];
  }
  __syncthreads();

  // ---- Phase 2: entmax15, register-resident rows (one wave per 2 rows) --
  for (int rr = 0; rr < 2; ++rr) {
    float* row = E + (2 * wid + rr) * ESTRIDE;
    float zr[NN / 32];
#pragma unroll
    for (int j = 0; j < NN / 32; ++j) zr[j] = row[lane + 32 * j];
    float mx = -3.0e38f;
#pragma unroll
    for (int j = 0; j < NN / 32; ++j) mx = fmaxf(mx, zr[j]);
    mx = wave_max(mx);
    float lo = mx - 1.0f, hi = mx;           // f(lo)>=1 > f(hi)=0
    for (int it = 0; it < 28; ++it) {
      const float mid = 0.5f * (lo + hi);
      float s = 0.0f;
#pragma unroll
      for (int j = 0; j < NN / 32; ++j) {
        float d = fmaxf(zr[j] - mid, 0.0f);
        s += d * d;
      }
      s = wave_sum(s);
      if (s >= 1.0f) lo = mid; else hi = mid;
    }
    const float tau = 0.5f * (lo + hi);
    // write p back packed f16 in place (row is private to this wave)
    _Float16* prow = (_Float16*)row;
#pragma unroll
    for (int j = 0; j < NN / 32; ++j) {
      float d = fmaxf(zr[j] - tau, 0.0f);
      prow[lane + 32 * j] = (_Float16)(d * d);   // p = (z - tau)^2
    }
  }
  __syncthreads();

  // ---- Phase 3: O = attn * V (A straight from packed-f16 LDS rows) ------
  const int ct = wid & 3;        // output col tile (16 of DK=64)
  const int khh = wid >> 2;      // which half of the 2048 K-range
  const _Float16* vrow = vtbuf + (bh * DKC + ct * 16 + ln) * NN;   // Vt row = B col
  const _Float16* parow = (const _Float16*)(E + (size_t)ln * ESTRIDE); // A row M=ln
  v8f acc = {};
  for (int c = 0; c < 32; ++c) {
    const int k0 = khh * 1024 + c * 32;
    v16h a = load_a16(parow, k0, hiL);            // 2x ds_load_b128
    v16h Bm = load_b16(vrow, k0, hiL);            // global_load_b128 x2
    acc = __builtin_amdgcn_wmma_f32_16x16x32_f16(false, a, false, Bm,
                                                 (short)0, acc, false, false);
  }
#pragma unroll
  for (int i = 0; i < 8; ++i)
    scr[wid * 256 + (i + 8 * hiL) * 16 + ln] = acc[i];
  __syncthreads();
  if (wid < 4) {
#pragma unroll
    for (int i = 0; i < 8; ++i) {
      const int idx = (i + 8 * hiL) * 16 + ln;
      const float v = scr[wid * 256 + idx] + scr[(wid + 4) * 256 + idx];
      const int n = ntile * 16 + i + 8 * hiL;
      const int cch = h * 64 + wid * 16 + ln;
      obuf[((size_t)b * NN + n) * CC + cch] = v;
    }
  }
}

// ---------------------------------------------------------------------------
// Kernel 3: LayerNorm (unbiased std, eps on std) + residual. One row/block.
// Single fused reduction pass: sum and sum-of-squares together.
// ---------------------------------------------------------------------------
__global__ void __launch_bounds__(256) hf_ln(
    const float* __restrict__ obuf, const float* __restrict__ xt,
    const float* __restrict__ lna, const float* __restrict__ lnb,
    float* __restrict__ out) {
  __shared__ float redس[16];
  // (avoid exotic identifiers)
  __shared__ float red[16];
  const int n = blockIdx.x, b = blockIdx.y;
  const int c = threadIdx.x, lane = c & 31, wid = c >> 5;
  const size_t idx = ((size_t)b * NN + n) * CC + c;
  const float v = obuf[idx];
  float s = wave_sum(v);
  float s2 = wave_sum(v * v);
  if (lane == 0) { red[wid] = s; red[8 + wid] = s2; }
  __syncthreads();
  float tot = 0.0f, tot2 = 0.0f;
#pragma unroll
  for (int i = 0; i < 8; ++i) { tot += red[i]; tot2 += red[8 + i]; }
  const float mean = tot * (1.0f / CC);
  // unbiased variance: (sum(x^2) - C*mean^2) / (C-1)
  const float var = fmaxf(tot2 - (float)CC * mean * mean, 0.0f) * (1.0f / (CC - 1));
  const float stdv = sqrtf(var);
  out[idx] = lna[c] * (v - mean) / (stdv + 1e-6f) + lnb[c] + xt[idx];
  (void)redس;
}

// ---------------------------------------------------------------------------
extern "C" void kernel_launch(void* const* d_in, const int* in_sizes, int n_in,
                              void* d_out, int out_size, void* d_ws, size_t ws_size,
                              hipStream_t stream) {
  (void)in_sizes; (void)n_in; (void)out_size; (void)ws_size;
  const float* x   = (const float*)d_in[0];
  const float* wq  = (const float*)d_in[1];
  const float* bq  = (const float*)d_in[2];
  const float* wk  = (const float*)d_in[3];
  const float* bk  = (const float*)d_in[4];
  const float* wv  = (const float*)d_in[5];
  const float* bv  = (const float*)d_in[6];
  const float* lna = (const float*)d_in[7];
  const float* lnb = (const float*)d_in[8];
  float* out = (float*)d_out;

  char* ws = (char*)d_ws;
  size_t off = 0;
  auto take = [&](size_t bytes) {
    char* p = ws + off;
    off += (bytes + 255) & ~(size_t)255;
    return p;
  };
  float*    xt    = (float*)take((size_t)BB * NN * CC * 4);
  _Float16* xh    = (_Float16*)take((size_t)BB * NN * CC * 2);
  _Float16* wqh   = (_Float16*)take((size_t)CC * CC * 2);
  _Float16* wkh   = (_Float16*)take((size_t)CC * CC * 2);
  _Float16* wvh   = (_Float16*)take((size_t)CC * CC * 2);
  _Float16* qbuf  = (_Float16*)take((size_t)BB * HH * NN * DKC * 2);
  _Float16* kbuf  = (_Float16*)take((size_t)BB * HH * NN * DKC * 2);
  _Float16* vtbuf = (_Float16*)take((size_t)BB * HH * DKC * NN * 2);
  float*    obuf  = (float*)take((size_t)BB * NN * CC * 4);

  hf_transpose<<<dim3(NN / 32, CC / 32, BB), dim3(32, 8), 0, stream>>>(x, xt, xh);
  hf_cvt_f16<<<(CC * CC + 255) / 256, 256, 0, stream>>>(wq, wqh, CC * CC);
  hf_cvt_f16<<<(CC * CC + 255) / 256, 256, 0, stream>>>(wk, wkh, CC * CC);
  hf_cvt_f16<<<(CC * CC + 255) / 256, 256, 0, stream>>>(wv, wvh, CC * CC);

  hf_qkv<<<dim3(NN / 16, BB), 256, 0, stream>>>(xh, wqh, wkh, wvh, bq, bk, bv,
                                                qbuf, kbuf, vtbuf);

  const size_t smem = (size_t)(16 * ESTRIDE + 8 * 256) * sizeof(float); // ~136.5 KB
  hf_attn<<<dim3(NN / 16, HH, BB), 256, smem, stream>>>(qbuf, kbuf, vtbuf, obuf);

  hf_ln<<<dim3(NN, BB), 256, 0, stream>>>(obuf, xt, lna, lnb, out);
}
